// Neural_ODE_39453569581646
// MI455X (gfx1250) — matrix operand order repaired and checked by reading, not compiled, not run on hardware
//
#include <hip/hip_runtime.h>

typedef _Float16 v8h  __attribute__((ext_vector_type(8)));
typedef _Float16 v16h __attribute__((ext_vector_type(16)));
typedef float    v8f  __attribute__((ext_vector_type(8)));

#define MB      64      // batch rows per block
#define STRIDE  1032    // halves per LDS row (1024 + 8 pad, staggers banks)
#define HDIM    1024
#define T_OUT   61
#define L_IN    80
#define K0PAD   96      // layer-0 K padded to multiple of 32
#define NTHREADS 1024   // 32 waves -> 8 waves per SIMD32

// ---------------------------------------------------------------------------
// Convert W (f32, row-major K x N) -> Wt (f16, row-major N x Kpad, zero pad)
// ---------------------------------------------------------------------------
__global__ __launch_bounds__(256)
void convert_wt(const float* __restrict__ W, _Float16* __restrict__ Wt,
                int K, int Kpad, int N) {
  int idx = blockIdx.x * blockDim.x + threadIdx.x;
  if (idx >= N * Kpad) return;
  int n = idx / Kpad;
  int k = idx - n * Kpad;
  float v = (k < K) ? W[(size_t)k * N + n] : 0.0f;
  Wt[(size_t)n * Kpad + k] = (_Float16)v;
}

// ---------------------------------------------------------------------------
// One MLP layer: bufOut = leaky_relu(bufIn @ Wt^T + bias), KD compile-time
// so B-tile offsets fold into the global_load instruction offset.
// Each wave: 1 M-tile (mt), 8 N-tiles starting at nbase0 (64 acc VGPRs --
// fits the ~128-VGPR cap at 8 waves/SIMD without spilling).
// ---------------------------------------------------------------------------
template<int KD>
__device__ __forceinline__ void mlp_layer(const _Float16* __restrict__ bufIn,
                                          _Float16* __restrict__ bufOut,
                                          const _Float16* __restrict__ Wt,
                                          const float* __restrict__ bias,
                                          int mt, int nbase0, int laneHi, int lmod) {
  v8f acc[8];
#pragma unroll
  for (int j = 0; j < 8; ++j) {
    float bv = bias[nbase0 + j * 16 + lmod];
    v8f t = {bv, bv, bv, bv, bv, bv, bv, bv};
    acc[j] = t;
  }

  // A fragment base (16x32 f16): lanes 0-15: row=lane, K = {0..7,16..23}
  //                              lanes16-31: row=lane-16, K = {8..15,24..31}
  const _Float16* ap = bufIn + (mt * 16 + lmod) * STRIDE + (laneHi ? 8 : 0);
  // B fragment base (32x16 f16): lane holds column n, K = {0..7,16..23} / {8..15,24..31}
  const _Float16* bp = Wt + (size_t)(nbase0 + lmod) * KD + (laneHi ? 8 : 0);

#pragma unroll 1
  for (int ks = 0; ks < KD / 32; ++ks) {
    v8h alo = *(const v8h*)(ap);
    v8h ahi = *(const v8h*)(ap + 16);
    v16h a = __builtin_shufflevector(alo, ahi,
               0,1,2,3,4,5,6,7,8,9,10,11,12,13,14,15);

    // phase 1: issue all 8 B-fragment loads (constant offsets -> one clause)
    v16h bf[8];
#pragma unroll
    for (int j = 0; j < 8; ++j) {
      const _Float16* p = bp + j * (16 * KD);
      v8h blo = *(const v8h*)(p);
      v8h bhi = *(const v8h*)(p + 16);
      bf[j] = __builtin_shufflevector(blo, bhi,
                0,1,2,3,4,5,6,7,8,9,10,11,12,13,14,15);
    }
    // phase 2: 8 back-to-back WMMAs on independent accumulators
#pragma unroll
    for (int j = 0; j < 8; ++j) {
      acc[j] = __builtin_amdgcn_wmma_f32_16x16x32_f16(
                 false, a, false, bf[j], (short)0, acc[j], false, false);
    }
    ap += 32;
    bp += 32;
  }

  // leaky-ReLU on f32 accumulators, convert to f16, store (D layout:
  // lane holds column n; VGPR e holds row e (+8 for lanes 16-31))
#pragma unroll
  for (int j = 0; j < 8; ++j) {
    const int col = nbase0 + j * 16 + lmod;
#pragma unroll
    for (int e = 0; e < 8; ++e) {
      float x = acc[j][e];
      x = (x > 0.0f) ? x : 0.01f * x;
      const int mrow = mt * 16 + e + laneHi * 8;
      bufOut[mrow * STRIDE + col] = (_Float16)x;
    }
  }
}

// ---------------------------------------------------------------------------
// Fused MLP (WMMA f16->f32) + final 1024->3 projection + RK4 SEIR integration
// ---------------------------------------------------------------------------
__global__ __launch_bounds__(NTHREADS)
void seir_mlp(const float* __restrict__ xx,
              const _Float16* __restrict__ wts,
              const float* __restrict__ b0, const float* __restrict__ b1,
              const float* __restrict__ b2, const float* __restrict__ b3,
              const float* __restrict__ b4, const float* __restrict__ b5,
              const float* __restrict__ W6, const float* __restrict__ b6,
              float* __restrict__ out) {
  __shared__ __align__(16) _Float16 smem[2 * MB * STRIDE];  // ~258 KB ping-pong
  __shared__ float bgs[MB][3];

  const int tid     = threadIdx.x;
  const int lane    = tid & 31;
  const int w       = tid >> 5;          // wave id, 0..31
  const int rowbase = blockIdx.x * MB;

  // ---- stage input rows (80 f32 -> f16, zero-pad to 96) into buffer 0 ----
  for (int idx = tid; idx < MB * K0PAD; idx += NTHREADS) {
    int r = idx / K0PAD;
    int k = idx - r * K0PAD;
    float v = (k < L_IN) ? xx[(size_t)(rowbase + r) * L_IN + k] : 0.0f;
    smem[r * STRIDE + k] = (_Float16)v;
  }
  __syncthreads();

  const int mt     = w & 3;              // M-tile (0..3)
  const int ngrp   = w >> 2;             // 0..7 -> which 128-col span
  const int nbase0 = ngrp * 128;         // 8 N-tiles of 16 cols
  const int laneHi = (lane >> 4) & 1;
  const int lmod   = lane & 15;

  _Float16* bufA = smem;
  _Float16* bufB = smem + MB * STRIDE;

  // layer 0 (K = 96, zero-padded)
  mlp_layer<K0PAD>(bufA, bufB, wts, b0, mt, nbase0, laneHi, lmod);
  __syncthreads();

  // layers 1..5 (K = 1024)
  const float* biases[5] = {b1, b2, b3, b4, b5};
  int cur = 1;
  for (int l = 0; l < 5; ++l) {
    const _Float16* Wt = wts + (size_t)HDIM * K0PAD + (size_t)l * HDIM * HDIM;
    const _Float16* bin  = (cur == 1) ? bufB : bufA;
    _Float16*       bout = (cur == 1) ? bufA : bufB;
    mlp_layer<HDIM>(bin, bout, Wt, biases[l], mt, nbase0, laneHi, lmod);
    __syncthreads();
    cur = 1 - cur;
  }
  // after 6 layers, activations live in: cur==1 -> bufB, cur==0 -> bufA
  const _Float16* hbuf = (cur == 1) ? bufB : bufA;

  // ---- final projection: out3 = h @ W6 + b6 (16 threads per row) ----
  {
    const int r = tid >> 4, sub = tid & 15;
    float a0 = 0.f, a1 = 0.f, a2 = 0.f;
    const int kb = sub * 64;
    for (int k = kb; k < kb + 64; ++k) {
      float hv = (float)hbuf[r * STRIDE + k];
      a0 += hv * W6[k * 3 + 0];
      a1 += hv * W6[k * 3 + 1];
      a2 += hv * W6[k * 3 + 2];
    }
    for (int off = 8; off > 0; off >>= 1) {
      a0 += __shfl_down(a0, off, 16);
      a1 += __shfl_down(a1, off, 16);
      a2 += __shfl_down(a2, off, 16);
    }
    if (sub == 0) {
      bgs[r][0] = a0 + b6[0];
      bgs[r][1] = a1 + b6[1];
      bgs[r][2] = a2 + b6[2];
    }
  }
  __syncthreads();

  // ---- RK4 SEIR integration, one thread per row ----
  if (tid < MB) {
    const int r = tid;
    const float beta  = bgs[r][0];
    const float gamma = bgs[r][1];
    const float sigma = bgs[r][2];
    const float* x0 = xx + (size_t)(rowbase + r) * L_IN;
    float S = x0[0], E = x0[1], I = x0[2], R = x0[3];
    float4* o = (float4*)(out + (size_t)(rowbase + r) * T_OUT * 4);
    o[0] = make_float4(S, E, I, R);
#define DERIV(S_, E_, I_, dS, dE, dI, dR)            \
    { float SI = beta * (S_) * (I_);                 \
      dS = -SI; dE = SI - sigma * (E_);              \
      dI = sigma * (E_) - gamma * (I_);              \
      dR = gamma * (I_); }
    for (int t = 1; t < T_OUT; ++t) {
      float k1S,k1E,k1I,k1R, k2S,k2E,k2I,k2R, k3S,k3E,k3I,k3R, k4S,k4E,k4I,k4R;
      DERIV(S, E, I, k1S, k1E, k1I, k1R);
      DERIV(S + 0.5f*k1S, E + 0.5f*k1E, I + 0.5f*k1I, k2S, k2E, k2I, k2R);
      DERIV(S + 0.5f*k2S, E + 0.5f*k2E, I + 0.5f*k2I, k3S, k3E, k3I, k3R);
      DERIV(S + k3S, E + k3E, I + k3I, k4S, k4E, k4I, k4R);
      const float c = 1.0f / 6.0f;
      S += (k1S + 2.f*k2S + 2.f*k3S + k4S) * c;
      E += (k1E + 2.f*k2E + 2.f*k3E + k4E) * c;
      I += (k1I + 2.f*k2I + 2.f*k3I + k4I) * c;
      R += (k1R + 2.f*k2R + 2.f*k3R + k4R) * c;
      o[t] = make_float4(S, E, I, R);
    }
#undef DERIV
  }
}

// ---------------------------------------------------------------------------
extern "C" void kernel_launch(void* const* d_in, const int* in_sizes, int n_in,
                              void* d_out, int out_size, void* d_ws, size_t ws_size,
                              hipStream_t stream) {
  const float* xx = (const float*)d_in[0];
  const float* W[7];
  const float* b[7];
  for (int i = 0; i < 7; ++i) {
    W[i] = (const float*)d_in[1 + 2 * i];
    b[i] = (const float*)d_in[2 + 2 * i];
  }
  _Float16* wts = (_Float16*)d_ws;

  // layer 0: 80x1024 -> f16 [1024][96] (zero-padded)
  {
    int total = HDIM * K0PAD;
    convert_wt<<<(total + 255) / 256, 256, 0, stream>>>(W[0], wts, L_IN, K0PAD, HDIM);
  }
  // layers 1..5: 1024x1024 -> f16 [1024][1024]
  for (int l = 1; l < 6; ++l) {
    _Float16* dst = wts + (size_t)HDIM * K0PAD + (size_t)(l - 1) * HDIM * HDIM;
    int total = HDIM * HDIM;
    convert_wt<<<(total + 255) / 256, 256, 0, stream>>>(W[l], dst, HDIM, HDIM, HDIM);
  }

  const int nblocks = 65536 / MB;
  seir_mlp<<<nblocks, NTHREADS, 0, stream>>>(
      xx, wts, b[0], b[1], b[2], b[3], b[4], b[5], W[6], b[6], (float*)d_out);
}
